// AttnBlock_57896159150484
// MI455X (gfx1250) — matrix operand order repaired and checked by reading, not compiled, not run on hardware
//
#include <hip/hip_runtime.h>
#include <hip/hip_bf16.h>

typedef __bf16 bf16_t;
typedef __bf16 bf16x8  __attribute__((ext_vector_type(8)));
typedef __bf16 bf16x16 __attribute__((ext_vector_type(16)));
typedef float  v8f     __attribute__((ext_vector_type(8)));

static constexpr int BATCH = 4;
static constexpr int C     = 512;
static constexpr int N     = 4096;     // H*W
static constexpr int K3    = 1536;     // 3*C
static constexpr int G     = 32;       // groups
static constexpr float SCALE = 0.04419417382415922f; // 512^-0.5

// ---------------- workspace layout (bytes) ----------------
static constexpr size_t SZ_BNC  = (size_t)BATCH * N * C * 2;     // 16 MiB bf16 tensor
static constexpr size_t OFF_HT  = 0;                             // hT   [B][N][C]
static constexpr size_t OFF_QT  = OFF_HT + SZ_BNC;               // qT   [B][N][C]
static constexpr size_t OFF_KT  = OFF_QT + SZ_BNC;               // kT   [B][N][C]
static constexpr size_t OFF_V   = OFF_KT + SZ_BNC;               // v    [B][C][N]
static constexpr size_t OFF_OT  = OFF_V  + SZ_BNC;               // oT   [B][N][C]
static constexpr size_t OFF_WQ  = OFF_OT + SZ_BNC;               // wq   [1536][512] bf16
static constexpr size_t OFF_WP  = OFF_WQ + (size_t)K3 * C * 2;   // wp   [512][512] bf16
static constexpr size_t OFF_S   = OFF_WP + (size_t)C * C * 2;    // S    [N][N] f32 (1 batch)
static constexpr size_t OFF_P   = OFF_S  + (size_t)N * N * 4;    // P    [N][N] bf16 (1 batch)

// ---------------- WMMA fragment loaders ----------------
// A fragment (16x32 bf16): lane L holds row (L&15); lanes 0-15: K {kc..kc+7, kc+16..kc+23},
// lanes 16-31: K {kc+8..kc+15, kc+24..kc+31}  (ISA 7.12.2, 16-bit A 16x32)
__device__ __forceinline__ bf16x16 load_frag_a(const bf16_t* __restrict__ base, int ld, int kc, int lane) {
    int row  = lane & 15;
    int koff = kc + ((lane & 16) ? 8 : 0);
    const bf16_t* p = base + (size_t)row * ld + koff;
    bf16x8 lo = *(const bf16x8*)(p);
    bf16x8 hi = *(const bf16x8*)(p + 16);
    bf16x16 r;
#pragma unroll
    for (int i = 0; i < 8; ++i) { r[i] = lo[i]; r[i + 8] = hi[i]; }
    return r;
}
// B fragment (32x16 bf16) loaded from B^T row-major (row = column n, contiguous K):
// lane L holds column (L&15); lanes 0-15: K {kc..kc+7, kc+16..kc+23}; lanes 16-31: K {kc+8..kc+15, kc+24..kc+31}.
__device__ __forceinline__ bf16x16 load_frag_b(const bf16_t* __restrict__ baseT, int ld, int kc, int lane) {
    int col  = lane & 15;
    int koff = kc + ((lane & 16) ? 8 : 0);
    const bf16_t* p = baseT + (size_t)col * ld + koff;
    bf16x8 lo = *(const bf16x8*)(p);
    bf16x8 hi = *(const bf16x8*)(p + 16);
    bf16x16 r;
#pragma unroll
    for (int i = 0; i < 8; ++i) { r[i] = lo[i]; r[i + 8] = hi[i]; }
    return r;
}

// ---------------- software-pipelined 32x64 per-wave GEMM core ----------------
// A: row-major, wave's 32-row strip (2 m-tiles). B^T: row-major, wave's 64-col strip (4 n-tiles).
// KDIM must be a power of two and a multiple of 64.
template <int KDIM>
__device__ __forceinline__ void gemm_core(const bf16_t* __restrict__ aB, int lda,
                                          const bf16_t* __restrict__ btB, int ldb,
                                          v8f acc[2][4], int lane) {
    bf16x16 a0[2], b0[4], a1[2], b1[4];
#pragma unroll
    for (int i = 0; i < 2; ++i) a0[i] = load_frag_a(aB + (size_t)i * 16 * lda, lda, 0, lane);
#pragma unroll
    for (int j = 0; j < 4; ++j) b0[j] = load_frag_b(btB + (size_t)j * 16 * ldb, ldb, 0, lane);
    for (int kc = 0; kc < KDIM; kc += 64) {
        // prefetch chunk kc+32 while set0 (kc) computes
#pragma unroll
        for (int i = 0; i < 2; ++i) a1[i] = load_frag_a(aB + (size_t)i * 16 * lda, lda, kc + 32, lane);
#pragma unroll
        for (int j = 0; j < 4; ++j) b1[j] = load_frag_b(btB + (size_t)j * 16 * ldb, ldb, kc + 32, lane);
#pragma unroll
        for (int i = 0; i < 2; ++i)
#pragma unroll
            for (int j = 0; j < 4; ++j)
                acc[i][j] = __builtin_amdgcn_wmma_f32_16x16x32_bf16(false, a0[i], false, b0[j], (short)0, acc[i][j], false, false);
        // prefetch chunk kc+64 (wraps harmlessly to 0 on the final iteration)
        const int kn = (kc + 64) & (KDIM - 1);
#pragma unroll
        for (int i = 0; i < 2; ++i) a0[i] = load_frag_a(aB + (size_t)i * 16 * lda, lda, kn, lane);
#pragma unroll
        for (int j = 0; j < 4; ++j) b0[j] = load_frag_b(btB + (size_t)j * 16 * ldb, ldb, kn, lane);
#pragma unroll
        for (int i = 0; i < 2; ++i)
#pragma unroll
            for (int j = 0; j < 4; ++j)
                acc[i][j] = __builtin_amdgcn_wmma_f32_16x16x32_bf16(false, a1[i], false, b1[j], (short)0, acc[i][j], false, false);
    }
}

__device__ __forceinline__ void zero_acc(v8f acc[2][4]) {
#pragma unroll
    for (int i = 0; i < 2; ++i)
#pragma unroll
        for (int j = 0; j < 4; ++j)
#pragma unroll
            for (int e = 0; e < 8; ++e) acc[i][j][e] = 0.f;
}

__device__ __forceinline__ float wave_red_sum(float v) {
#pragma unroll
    for (int o = 16; o > 0; o >>= 1) v += __shfl_xor(v, o, 32);
    return v;
}
__device__ __forceinline__ float wave_red_max(float v) {
#pragma unroll
    for (int o = 16; o > 0; o >>= 1) v = fmaxf(v, __shfl_xor(v, o, 32));
    return v;
}

// ---------------- kernel 0: weight fp32 -> bf16 ----------------
__global__ __launch_bounds__(256) void convw_kernel(const float* __restrict__ qkv_w,
                                                    const float* __restrict__ proj_w,
                                                    bf16_t* __restrict__ wq,
                                                    bf16_t* __restrict__ wp) {
    int id = blockIdx.x * 256 + threadIdx.x;
    const int n1 = K3 * C;
    if (id < n1) wq[id] = (bf16_t)qkv_w[id];
    int id2 = id - n1;
    if (id2 >= 0 && id2 < C * C) wp[id2] = (bf16_t)proj_w[id2];
}

// ---------------- kernel 1: GroupNorm -> hT (bf16, [B][N][C]) ----------------
__global__ __launch_bounds__(256) void gnorm_kernel(const float* __restrict__ x,
                                                    const float* __restrict__ gamma,
                                                    const float* __restrict__ beta,
                                                    bf16_t* __restrict__ hT) {
    __shared__ float red[16];
    const int b = blockIdx.y, g = blockIdx.x, tid = threadIdx.x;
    const float* base = x + ((size_t)b * C + (size_t)g * 16) * N;  // 16 channels contiguous
    const int NV4 = 16 * N / 4;  // 16384 float4s in this group
    float s = 0.f, q = 0.f;
    for (int i = tid; i < NV4; i += 256) {
        float4 v = ((const float4*)base)[i];
        s += v.x + v.y + v.z + v.w;
        q += v.x * v.x + v.y * v.y + v.z * v.z + v.w * v.w;
    }
    s = wave_red_sum(s); q = wave_red_sum(q);
    if ((tid & 31) == 0) { red[tid >> 5] = s; red[8 + (tid >> 5)] = q; }
    __syncthreads();
    float S = 0.f, Q = 0.f;
#pragma unroll
    for (int w = 0; w < 8; ++w) { S += red[w]; Q += red[8 + w]; }
    const float invM = 1.0f / 65536.0f;
    float mean = S * invM;
    float var  = Q * invM - mean * mean;
    float inv  = rsqrtf(var + 1e-5f);
    for (int i = tid; i < NV4; i += 256) {
        float4 v = ((const float4*)base)[i];
        int idx = i * 4;
        int cl = idx >> 12;            // channel within group (4096 per channel)
        int n  = idx & (N - 1);        // n..n+3 share the same channel
        int c  = g * 16 + cl;
        float ga = gamma[c], be = beta[c];
        bf16_t* dst = hT + ((size_t)b * N + n) * C + c;
        dst[0 * C] = (bf16_t)((v.x - mean) * inv * ga + be);
        dst[1 * C] = (bf16_t)((v.y - mean) * inv * ga + be);
        dst[2 * C] = (bf16_t)((v.z - mean) * inv * ga + be);
        dst[3 * C] = (bf16_t)((v.w - mean) * inv * ga + be);
    }
}

// ---------------- kernel 2: QKV GEMM  qkv[o][n] = Wq[o][:]·hT[n][:] + b ----------------
// block tile 64(o) x 128(n); wave tile 32x64
__global__ __launch_bounds__(128) void qkv_gemm_kernel(const bf16_t* __restrict__ wq,
                                                       const bf16_t* __restrict__ hT,
                                                       const float* __restrict__ qkv_b,
                                                       bf16_t* __restrict__ qT,
                                                       bf16_t* __restrict__ kT,
                                                       bf16_t* __restrict__ vM) {
    const int lane = threadIdx.x & 31, wv = threadIdx.x >> 5;
    const int wm = wv & 1, wn = wv >> 1;
    const int b = blockIdx.z;
    const int nBase = blockIdx.x * 128 + wn * 64;
    const int oBase = blockIdx.y * 64 + wm * 32;
    const bf16_t* aB  = wq + (size_t)oBase * C;
    const bf16_t* btB = hT + ((size_t)b * N + nBase) * C;
    v8f acc[2][4];
    zero_acc(acc);
    gemm_core<C>(aB, C, btB, C, acc, lane);
    const int nCol = lane & 15, rOff = (lane >> 4) << 3;
#pragma unroll
    for (int i = 0; i < 2; ++i)
#pragma unroll
        for (int t = 0; t < 4; ++t) {
            int n = nBase + t * 16 + nCol;
#pragma unroll
            for (int r = 0; r < 8; ++r) {
                int o = oBase + i * 16 + r + rOff;
                float val = acc[i][t][r] + qkv_b[o];
                if (o < C)            qT[((size_t)b * N + n) * C + o]           = (bf16_t)val;
                else if (o < 2 * C)   kT[((size_t)b * N + n) * C + (o - C)]     = (bf16_t)val;
                else                  vM[((size_t)b * C + (o - 2 * C)) * N + n] = (bf16_t)val;
            }
        }
}

// ---------------- kernel 3: S = Q^T K  (per batch, fp32 out) ----------------
__global__ __launch_bounds__(128) void qk_gemm_kernel(const bf16_t* __restrict__ qT,
                                                      const bf16_t* __restrict__ kT,
                                                      float* __restrict__ S) {
    const int lane = threadIdx.x & 31, wv = threadIdx.x >> 5;
    const int wm = wv & 1, wn = wv >> 1;
    const int jBase = blockIdx.x * 128 + wn * 64;
    const int iBase = blockIdx.y * 64 + wm * 32;
    const bf16_t* aB  = qT + (size_t)iBase * C;
    const bf16_t* btB = kT + (size_t)jBase * C;
    v8f acc[2][4];
    zero_acc(acc);
    gemm_core<C>(aB, C, btB, C, acc, lane);
    const int nCol = lane & 15, rOff = (lane >> 4) << 3;
#pragma unroll
    for (int i = 0; i < 2; ++i)
#pragma unroll
        for (int t = 0; t < 4; ++t)
#pragma unroll
            for (int r = 0; r < 8; ++r)
                S[(size_t)(iBase + i * 16 + r + rOff) * N + jBase + t * 16 + nCol] = acc[i][t][r];
}

// ---------------- kernel 4: row softmax (scale folded), P bf16 ----------------
__global__ __launch_bounds__(256) void softmax_kernel(const float* __restrict__ S,
                                                      bf16_t* __restrict__ P) {
    __shared__ float red[8];
    const int row = blockIdx.x, tid = threadIdx.x;
    const float* s = S + (size_t)row * N;
    float m = -3.4e38f;
    for (int j = tid; j < N; j += 256) m = fmaxf(m, s[j]);
    m = wave_red_max(m);
    if ((tid & 31) == 0) red[tid >> 5] = m;
    __syncthreads();
#pragma unroll
    for (int w = 0; w < 8; ++w) m = fmaxf(m, red[w]);
    float sum = 0.f;
    for (int j = tid; j < N; j += 256) sum += __expf(SCALE * (s[j] - m));
    sum = wave_red_sum(sum);
    __syncthreads();
    if ((tid & 31) == 0) red[tid >> 5] = sum;
    __syncthreads();
    float tot = 0.f;
#pragma unroll
    for (int w = 0; w < 8; ++w) tot += red[w];
    float inv = 1.0f / tot;
    bf16_t* p = P + (size_t)row * N;
    for (int j = tid; j < N; j += 256) p[j] = (bf16_t)(__expf(SCALE * (s[j] - m)) * inv);
}

// ---------------- kernel 5: O = V · P^T  (per batch), store O^T [i][c] ----------------
__global__ __launch_bounds__(128) void av_gemm_kernel(const bf16_t* __restrict__ vM,
                                                      const bf16_t* __restrict__ P,
                                                      bf16_t* __restrict__ oT) {
    const int lane = threadIdx.x & 31, wv = threadIdx.x >> 5;
    const int wm = wv & 1, wn = wv >> 1;
    const int iBase = blockIdx.x * 128 + wn * 64;
    const int cBase = blockIdx.y * 64 + wm * 32;
    const bf16_t* aB  = vM + (size_t)cBase * N;   // A[m=c][k=j]
    const bf16_t* btB = P  + (size_t)iBase * N;   // B^T row = i, contiguous j
    v8f acc[2][4];
    zero_acc(acc);
    gemm_core<N>(aB, N, btB, N, acc, lane);
    const int nCol = lane & 15, rOff = (lane >> 4) << 3;
#pragma unroll
    for (int i = 0; i < 2; ++i)
#pragma unroll
        for (int t = 0; t < 4; ++t) {
            int n = iBase + t * 16 + nCol;  // query index i
#pragma unroll
            for (int r = 0; r < 8; ++r) {
                int m = cBase + i * 16 + r + rOff;  // channel c
                oT[(size_t)n * C + m] = (bf16_t)acc[i][t][r];
            }
        }
}

// ---------------- kernel 6: proj GEMM + bias + residual ----------------
__global__ __launch_bounds__(128) void proj_gemm_kernel(const bf16_t* __restrict__ wp,
                                                        const bf16_t* __restrict__ oT,
                                                        const float* __restrict__ proj_b,
                                                        const float* __restrict__ x,
                                                        float* __restrict__ out) {
    const int lane = threadIdx.x & 31, wv = threadIdx.x >> 5;
    const int wm = wv & 1, wn = wv >> 1;
    const int b = blockIdx.z;
    const int nBase = blockIdx.x * 128 + wn * 64;
    const int oBase = blockIdx.y * 64 + wm * 32;
    const bf16_t* aB  = wp + (size_t)oBase * C;
    const bf16_t* btB = oT + ((size_t)b * N + nBase) * C;
    v8f acc[2][4];
    zero_acc(acc);
    gemm_core<C>(aB, C, btB, C, acc, lane);
    const int nCol = lane & 15, rOff = (lane >> 4) << 3;
#pragma unroll
    for (int i = 0; i < 2; ++i)
#pragma unroll
        for (int t = 0; t < 4; ++t) {
            int n = nBase + t * 16 + nCol;
#pragma unroll
            for (int r = 0; r < 8; ++r) {
                int m = oBase + i * 16 + r + rOff;
                size_t idx = ((size_t)b * C + m) * N + n;
                out[idx] = acc[i][t][r] + proj_b[m] + x[idx];
            }
        }
}

// ---------------- launcher ----------------
extern "C" void kernel_launch(void* const* d_in, const int* in_sizes, int n_in,
                              void* d_out, int out_size, void* d_ws, size_t ws_size,
                              hipStream_t stream) {
    const float* x      = (const float*)d_in[0];
    const float* gamma  = (const float*)d_in[1];
    const float* beta   = (const float*)d_in[2];
    const float* qkv_w  = (const float*)d_in[3];
    const float* qkv_b  = (const float*)d_in[4];
    const float* proj_w = (const float*)d_in[5];
    const float* proj_b = (const float*)d_in[6];
    float* out = (float*)d_out;
    char* ws = (char*)d_ws;

    bf16_t* hT = (bf16_t*)(ws + OFF_HT);
    bf16_t* qT = (bf16_t*)(ws + OFF_QT);
    bf16_t* kT = (bf16_t*)(ws + OFF_KT);
    bf16_t* vM = (bf16_t*)(ws + OFF_V);
    bf16_t* oT = (bf16_t*)(ws + OFF_OT);
    bf16_t* wq = (bf16_t*)(ws + OFF_WQ);
    bf16_t* wp = (bf16_t*)(ws + OFF_WP);
    float*  S  = (float*)(ws + OFF_S);
    bf16_t* P  = (bf16_t*)(ws + OFF_P);

    convw_kernel<<<(K3 * C + C * C + 255) / 256, 256, 0, stream>>>(qkv_w, proj_w, wq, wp);
    gnorm_kernel<<<dim3(G, BATCH), 256, 0, stream>>>(x, gamma, beta, hT);
    qkv_gemm_kernel<<<dim3(N / 128, K3 / 64, BATCH), 128, 0, stream>>>(wq, hT, qkv_b, qT, kT, vM);
    for (int b = 0; b < BATCH; ++b) {
        const size_t bo = (size_t)b * N * C;
        qk_gemm_kernel<<<dim3(N / 128, N / 64), 128, 0, stream>>>(qT + bo, kT + bo, S);
        softmax_kernel<<<N, 256, 0, stream>>>(S, P);
        av_gemm_kernel<<<dim3(N / 128, C / 64), 128, 0, stream>>>(vM + bo, P, oT + bo);
    }
    proj_gemm_kernel<<<dim3(N / 128, C / 64, BATCH), 128, 0, stream>>>(wp, oT, proj_b, x, out);
}
